// Decoder_20366734918129
// MI455X (gfx1250) — matrix operand order repaired and checked by reading, not compiled, not run on hardware
//
#include <hip/hip_runtime.h>
#include <hip/hip_bf16.h>
#include <stdint.h>

// Problem constants (from reference)
#define BQ    32
#define TENC  512
#define TDEC  32
#define EMBD  256
#define NVOC  32000
#define CHUNK 8
#define SOSTK 1

typedef __attribute__((ext_vector_type(16))) __bf16          v16bf;
typedef __attribute__((ext_vector_type(8)))  float           v8f;
typedef __attribute__((ext_vector_type(8)))  unsigned short  u16x8;
typedef __attribute__((ext_vector_type(16))) unsigned short  u16x16;

#if defined(__has_builtin)
#if __has_builtin(__builtin_amdgcn_global_load_async_to_lds_b32) && \
    __has_builtin(__builtin_amdgcn_s_wait_asynccnt)
#define USE_ASYNC_LDS 1
#endif
#if __has_builtin(__builtin_amdgcn_tanhf)
#define HW_TANH(x) __builtin_amdgcn_tanhf(x)
#elif __has_builtin(__builtin_amdgcn_tanh_f32)
#define HW_TANH(x) __builtin_amdgcn_tanh_f32(x)
#endif
#endif

typedef __attribute__((address_space(1))) int* gptr_i32;
typedef __attribute__((address_space(3))) int* lptr_i32;

static __device__ __forceinline__ unsigned short f2bf_bits(float f) {
  unsigned u = __builtin_bit_cast(unsigned, f);
  unsigned r = u + 0x7FFFu + ((u >> 16) & 1u);   // round-to-nearest-even
  return (unsigned short)(r >> 16);
}
static __device__ __forceinline__ float fast_tanh(float x) {
#if defined(HW_TANH)
  return HW_TANH(x);                              // v_tanh_f32 (CDNA5 TRANS op)
#else
  float e = __expf(2.0f * x);
  return 1.0f - 2.0f / (e + 1.0f);
#endif
}

// A fragment from bf16 memory: 8 contiguous K at p, 8 contiguous K at p+16.
static __device__ __forceinline__ v16bf frag_a(const unsigned short* p) {
  u16x8 a = *(const u16x8*)p;
  u16x8 b = *(const u16x8*)(p + 16);
  u16x16 t;
#pragma unroll
  for (int i = 0; i < 8; ++i) { t[i] = a[i]; t[8 + i] = b[i]; }
  return __builtin_bit_cast(v16bf, t);
}
// B fragment: 16 contiguous K values (32B aligned).
static __device__ __forceinline__ v16bf frag_b(const unsigned short* p) {
  u16x16 t = *(const u16x16*)p;
  return __builtin_bit_cast(v16bf, t);
}

// ---------------------------------------------------------------------------
// Generic WMMA GEMM:  C[M,N] = act( A[M,K] * W[N,K]^T + bias )
// A and W are bf16 (activations pre-converted once at their producer).
// One wave computes one 16x16 C tile; K fully unrolled in steps of 32 so the
// body is a pure global_load_b128 -> v_wmma_f32_16x16x32_bf16 stream.
// Fragment layouts follow CDNA5 ISA 7.12.2 (16-bit A 16x32 / B 32x16, f32 C).
// ---------------------------------------------------------------------------
template <int KD, bool BIAS, bool TANHACT, bool OUT_BF16, bool DUALOUT>
__global__ void gemm_wmma_kernel(const unsigned short* __restrict__ A, int lda,
                                 const unsigned short* __restrict__ W,
                                 const float* __restrict__ bias,
                                 void* __restrict__ Cv,
                                 unsigned short* __restrict__ Cbf,
                                 long ldc, long out_off,
                                 int Mtiles, int Ntiles) {
  int wg = (int)((blockIdx.x * blockDim.x + threadIdx.x) >> 5);
  if (wg >= Mtiles * Ntiles) return;               // wave-uniform guard
  int lane    = (int)(threadIdx.x & 31);
  int lane_hi = lane >> 4;                          // 0: lanes 0-15, 1: lanes 16-31
  int l16     = lane & 15;
  int mt = wg / Ntiles, nt = wg % Ntiles;
  int arow = mt * 16 + l16;                         // A row for this lane
  int ncol = nt * 16 + l16;                         // B column / C column

  const unsigned short* Arow = A + (size_t)arow * (size_t)lda;
  const unsigned short* Wrow = W + (size_t)ncol * (size_t)KD;
  __builtin_prefetch(Wrow, 0, 1);                   // global_prefetch for W row

  v8f c = {0.f, 0.f, 0.f, 0.f, 0.f, 0.f, 0.f, 0.f};

#pragma unroll
  for (int k0 = 0; k0 < KD; k0 += 32) {
    v16bf a = frag_a(Arow + k0 + lane_hi * 8);      // A: 8 contig K, then +16
    v16bf b = frag_a(Wrow + k0 + lane_hi * 8);      // B: 8 contig K, then +16
    c = __builtin_amdgcn_wmma_f32_16x16x32_bf16(false, a, false, b,
                                                (short)0, c, false, false);
  }

  float bv = BIAS ? bias[ncol] : 0.0f;
#pragma unroll
  for (int r = 0; r < 8; ++r) {
    long m = (long)mt * 16 + r + lane_hi * 8;       // C: M = r + 8*lane_hi
    float v = c[r] + bv;
    if (TANHACT) v = fast_tanh(v);
    if (OUT_BF16) ((unsigned short*)Cv)[m * ldc + ncol] = f2bf_bits(v);
    else          ((float*)Cv)[out_off + m * ldc + ncol] = v;
    if (DUALOUT)  Cbf[m * ldc + ncol] = f2bf_bits(v);
  }
}

// ---------------------------------------------------------------------------
// Small setup kernels
// ---------------------------------------------------------------------------
__global__ void cvt_bf16_kernel(const float* __restrict__ src,
                                unsigned short* __restrict__ dst, int n) {
  int i = (int)(blockIdx.x * blockDim.x + threadIdx.x);
  if (i < n) dst[i] = f2bf_bits(src[i]);
}

// Wcat[a][k] = k<256 ? W_ih[a][k] : W_hh[a][k-256];  bcat[a] = b_ih[a]+b_hh[a]
__global__ void build_wcat_kernel(const float* __restrict__ Wih, const float* __restrict__ Whh,
                                  const float* __restrict__ bih, const float* __restrict__ bhh,
                                  unsigned short* __restrict__ Wcat, float* __restrict__ bcat) {
  int idx = (int)(blockIdx.x * blockDim.x + threadIdx.x);   // 256*512
  int a = idx >> 9, k = idx & 511;
  float v = (k < 256) ? Wih[a * 256 + k] : Whh[a * 256 + (k - 256)];
  Wcat[idx] = f2bf_bits(v);
  if (idx < 256) bcat[idx] = bih[idx] + bhh[idx];
}

// Vcat[n][k] = n<256 ? Vm[n][k] : Vc[n-256][k]
__global__ void build_vcat_kernel(const float* __restrict__ Vm, const float* __restrict__ Vc,
                                  unsigned short* __restrict__ Vcat) {
  int idx = (int)(blockIdx.x * blockDim.x + threadIdx.x);   // 512*256
  int n = idx >> 8, k = idx & 255;
  float v = (n < 256) ? Vm[n * 256 + k] : Vc[(n - 256) * 256 + k];
  Vcat[idx] = f2bf_bits(v);
}

// weff = (g / ||vv||) * vv
__global__ void weff_kernel(const float* __restrict__ vv, const float* __restrict__ g,
                            float* __restrict__ weff) {
  __shared__ float red[256];
  int i = (int)threadIdx.x;
  float v = vv[i];
  red[i] = v * v;
  __syncthreads();
  for (int off = 128; off > 0; off >>= 1) {
    if (i < off) red[i] += red[i + off];
    __syncthreads();
  }
  weff[i] = (g[0] / sqrtf(red[0])) * v;
}

// alpha[:,0] = 1 else 0; h = h0
__global__ void init_kernel(float* __restrict__ alpha, const float* __restrict__ h0,
                            float* __restrict__ h) {
  int i = (int)(blockIdx.x * blockDim.x + threadIdx.x);     // BQ*TENC
  if (i < BQ * TENC) alpha[i] = ((i & (TENC - 1)) == 0) ? 1.0f : 0.0f;
  if (i < BQ * EMBD) h[i] = h0[i];
}

// xh[b][k] = k<256 ? emb[tok[b]][k] : h_prev[b][k-256]  (bf16)
__global__ void build_xh_kernel(const float* __restrict__ emb, const int* __restrict__ decin,
                                const float* __restrict__ h, unsigned short* __restrict__ xh,
                                int t) {
  int idx = (int)(blockIdx.x * blockDim.x + threadIdx.x);   // BQ*512
  int b = idx >> 9, k = idx & 511;
  float v;
  if (k < 256) {
    int tok = (t == 0) ? SOSTK : decin[b * TDEC + (t - 1)];
    v = emb[(size_t)tok * 256 + k];
  } else {
    v = h[b * 256 + (k - 256)];
  }
  xh[idx] = f2bf_bits(v);
}

// ---------------------------------------------------------------------------
// Energy: e[b,t] = sum_a weff[a]*tanh(encW[b,t,a] + hV[b,a]) + vb + r
// One wave32 per (b,t) row; both heads in one pass; hardware v_tanh_f32.
// ---------------------------------------------------------------------------
__global__ void energy_kernel(const float* __restrict__ encWm, const float* __restrict__ encWc,
                              const float* __restrict__ hV,
                              const float* __restrict__ weffm, const float* __restrict__ weffc,
                              const float* __restrict__ vbm, const float* __restrict__ rm,
                              const float* __restrict__ vbc, const float* __restrict__ rc,
                              float* __restrict__ e_m, float* __restrict__ e_c) {
  int wid  = (int)((blockIdx.x * blockDim.x + threadIdx.x) >> 5);
  int lane = (int)(threadIdx.x & 31);
  if (wid >= BQ * TENC) return;
  int b = wid >> 9;
  const float* pm = encWm + (size_t)wid * 256;
  const float* pc = encWc + (size_t)wid * 256;
  const float* hm = hV + b * 512;
  const float* hc = hV + b * 512 + 256;
  float sm = 0.0f, sc = 0.0f;
#pragma unroll
  for (int a = lane; a < 256; a += 32) {
    sm += weffm[a] * fast_tanh(pm[a] + hm[a]);
    sc += weffc[a] * fast_tanh(pc[a] + hc[a]);
  }
  for (int off = 16; off > 0; off >>= 1) {
    sm += __shfl_down(sm, off, 32);
    sc += __shfl_down(sc, off, 32);
  }
  if (lane == 0) {
    e_m[wid] = sm + vbm[0] + rm[0];
    e_c[wid] = sc + vbc[0] + rc[0];
  }
}

// ---------------------------------------------------------------------------
// Monotonic alpha + chunkwise beta. One block (512 threads) per batch row.
// LDS Hillis-Steele scans.
// ---------------------------------------------------------------------------
static __device__ __forceinline__ float block_scan(float v, float* buf) {
  int j = (int)threadIdx.x;
  __syncthreads();
  buf[j] = v;
  __syncthreads();
  for (int off = 1; off < TENC; off <<= 1) {
    float t = (j >= off) ? buf[j - off] : 0.0f;
    __syncthreads();
    buf[j] += t;
    __syncthreads();
  }
  return buf[j];
}
static __device__ __forceinline__ float block_max(float v, float* buf) {
  int j = (int)threadIdx.x;
  __syncthreads();
  buf[j] = v;
  __syncthreads();
  for (int off = TENC / 2; off > 0; off >>= 1) {
    if (j < off) buf[j] = fmaxf(buf[j], buf[j + off]);
    __syncthreads();
  }
  float m = buf[0];
  __syncthreads();
  return m;
}

__global__ void alphabeta_kernel(const float* __restrict__ e_m, const float* __restrict__ e_c,
                                 const float* __restrict__ noise_t,
                                 float* __restrict__ alpha, float* __restrict__ beta) {
  __shared__ float buf[TENC];
  int b = (int)blockIdx.x, j = (int)threadIdx.x;
  int idx = b * TENC + j;

  // p and exclusive safe cumprod of (1-p)
  float p   = 1.0f / (1.0f + __expf(-(e_m[idx] + noise_t[idx])));
  float omp = 1.0f - p;
  float lg  = __logf(fminf(fmaxf(omp, 1e-10f), 1.0f));
  float incl = block_scan(lg, buf);
  float cp   = __expf(incl - lg);                   // exclusive cumprod

  float aprev = alpha[idx];
  float q  = aprev / fminf(fmaxf(cp, 1e-10f), 1.0f);
  float cq = block_scan(q, buf);
  float anew = p * cp * cq;
  alpha[idx] = anew;

  // chunkwise beta
  float u    = e_c[idx];
  float umax = block_max(u, buf);
  float eu   = __expf(u - umax);
  block_scan(eu, buf);                              // buf = incl cumsum(eu)
  float ms1   = buf[j] - ((j >= CHUNK) ? buf[j - CHUNK] : 0.0f);
  float denom = fmaxf(ms1, 1e-10f);
  float r2    = anew / denom;
  block_scan(r2, buf);                              // buf = incl cumsum(r2)
  int hiidx = j + CHUNK - 1; if (hiidx > TENC - 1) hiidx = TENC - 1;
  float hi = buf[hiidx];
  float lo = (j > 0) ? buf[j - 1] : 0.0f;
  beta[idx] = eu * (hi - lo);
}

// ---------------------------------------------------------------------------
// context[b][e] = sum_t beta[b][t]*enc[b][t][e];  cat(bf16) = [context, h]
// beta staged in LDS via gfx1250 async global->LDS DMA when available.
// ---------------------------------------------------------------------------
__global__ void context_kernel(const float* __restrict__ beta, const float* __restrict__ enc,
                               const unsigned short* __restrict__ hbf,
                               unsigned short* __restrict__ catbf) {
  __shared__ float sb[TENC];
  int b = (int)blockIdx.x, e = (int)threadIdx.x;
#if defined(USE_ASYNC_LDS)
  __builtin_amdgcn_global_load_async_to_lds_b32(
      (gptr_i32)(beta + b * TENC + e),
      (lptr_i32)(&sb[e]), 0, 0);
  __builtin_amdgcn_global_load_async_to_lds_b32(
      (gptr_i32)(beta + b * TENC + 256 + e),
      (lptr_i32)(&sb[e + 256]), 0, 0);
  __builtin_amdgcn_s_wait_asynccnt(0);
#else
  sb[e]       = beta[b * TENC + e];
  sb[e + 256] = beta[b * TENC + e + 256];
#endif
  __syncthreads();
  const float* ep = enc + (size_t)b * TENC * 256 + e;
  float acc = 0.0f;
#pragma unroll 4
  for (int t = 0; t < TENC; ++t) acc += sb[t] * ep[(size_t)t * 256];
  catbf[b * 512 + e]       = f2bf_bits(acc);
  catbf[b * 512 + 256 + e] = hbf[b * 256 + e];
}

// ---------------------------------------------------------------------------
// Host orchestration
// ---------------------------------------------------------------------------
extern "C" void kernel_launch(void* const* d_in, const int* in_sizes, int n_in,
                              void* d_out, int out_size, void* d_ws, size_t ws_size,
                              hipStream_t stream) {
  const float* enc   = (const float*)d_in[0];
  const int*   decin = (const int*)d_in[1];
  const float* h0    = (const float*)d_in[2];
  const float* noise = (const float*)d_in[3];
  const float* emb   = (const float*)d_in[4];
  const float* Wih   = (const float*)d_in[5];
  const float* bih   = (const float*)d_in[6];
  const float* Whh   = (const float*)d_in[7];
  const float* bhh   = (const float*)d_in[8];
  const float* Wm    = (const float*)d_in[9];
  const float* Vm    = (const float*)d_in[10];
  const float* bm    = (const float*)d_in[11];
  const float* vvm   = (const float*)d_in[12];
  const float* gm    = (const float*)d_in[13];
  const float* vbm   = (const float*)d_in[14];
  const float* rm    = (const float*)d_in[15];
  const float* Wc    = (const float*)d_in[16];
  const float* Vc    = (const float*)d_in[17];
  const float* bc    = (const float*)d_in[18];
  const float* vvc   = (const float*)d_in[19];
  const float* gc    = (const float*)d_in[20];
  const float* vbc   = (const float*)d_in[21];
  const float* rc    = (const float*)d_in[22];
  const float* Wcomb = (const float*)d_in[23];
  const float* Wproj = (const float*)d_in[24];
  const float* bproj = (const float*)d_in[25];
  float* out = (float*)d_out;

  // workspace carve (256B aligned)
  char* wp = (char*)d_ws;
  auto carve = [&](size_t bytes) -> char* {
    char* p = wp;
    wp += (bytes + 255) & ~(size_t)255;
    return p;
  };
  float*          encWm    = (float*)carve((size_t)BQ * TENC * 256 * 4);
  float*          encWc    = (float*)carve((size_t)BQ * TENC * 256 * 4);
  unsigned short* Wproj_bf = (unsigned short*)carve((size_t)NVOC * 256 * 2);
  unsigned short* enc_bf   = (unsigned short*)carve((size_t)BQ * TENC * 256 * 2);
  unsigned short* Wm_bf    = (unsigned short*)carve(256 * 256 * 2);
  unsigned short* Wc_bf    = (unsigned short*)carve(256 * 256 * 2);
  unsigned short* Wcat_bf  = (unsigned short*)carve(256 * 512 * 2);
  unsigned short* Vcat_bf  = (unsigned short*)carve(512 * 256 * 2);
  unsigned short* Wcomb_bf = (unsigned short*)carve(256 * 512 * 2);
  float*          bcat     = (float*)carve(256 * 4);
  float*          weffm    = (float*)carve(256 * 4);
  float*          weffc    = (float*)carve(256 * 4);
  float*          hA       = (float*)carve(BQ * 256 * 4);
  float*          hB       = (float*)carve(BQ * 256 * 4);
  unsigned short* h_bf     = (unsigned short*)carve(BQ * 256 * 2);
  unsigned short* xh       = (unsigned short*)carve(BQ * 512 * 2);
  float*          hV       = (float*)carve(BQ * 512 * 4);
  float*          e_m      = (float*)carve(BQ * TENC * 4);
  float*          e_c      = (float*)carve(BQ * TENC * 4);
  float*          alpha    = (float*)carve(BQ * TENC * 4);
  float*          betab    = (float*)carve(BQ * TENC * 4);
  unsigned short* cat_bf   = (unsigned short*)carve(BQ * 512 * 2);
  unsigned short* attn_bf  = (unsigned short*)carve(BQ * 256 * 2);

  // --- one-time setup ---
  cvt_bf16_kernel<<<(BQ * TENC * 256) / 256, 256, 0, stream>>>(enc, enc_bf, BQ * TENC * 256);
  cvt_bf16_kernel<<<(256 * 256) / 256, 256, 0, stream>>>(Wm, Wm_bf, 256 * 256);
  cvt_bf16_kernel<<<(256 * 256) / 256, 256, 0, stream>>>(Wc, Wc_bf, 256 * 256);
  cvt_bf16_kernel<<<(256 * 512) / 256, 256, 0, stream>>>(Wcomb, Wcomb_bf, 256 * 512);
  cvt_bf16_kernel<<<(NVOC * 256) / 256, 256, 0, stream>>>(Wproj, Wproj_bf, NVOC * 256);
  build_wcat_kernel<<<(256 * 512) / 256, 256, 0, stream>>>(Wih, Whh, bih, bhh, Wcat_bf, bcat);
  build_vcat_kernel<<<(512 * 256) / 256, 256, 0, stream>>>(Vm, Vc, Vcat_bf);
  weff_kernel<<<1, 256, 0, stream>>>(vvm, gm, weffm);
  weff_kernel<<<1, 256, 0, stream>>>(vvc, gc, weffc);
  init_kernel<<<(BQ * TENC) / 256, 256, 0, stream>>>(alpha, h0, hA);

  // Precompute encW = enc @ W^T + b  (step-invariant; hoisted out of the scan)
  gemm_wmma_kernel<256, true, false, false, false><<<2048, 256, 0, stream>>>(
      enc_bf, 256, Wm_bf, bm, encWm, nullptr, 256, 0, (BQ * TENC) / 16, 16);
  gemm_wmma_kernel<256, true, false, false, false><<<2048, 256, 0, stream>>>(
      enc_bf, 256, Wc_bf, bc, encWc, nullptr, 256, 0, (BQ * TENC) / 16, 16);

  // --- sequential decode scan ---
  float* hprev = hA;
  float* hnext = hB;
  for (int t = 0; t < TDEC; ++t) {
    build_xh_kernel<<<(BQ * 512) / 256, 256, 0, stream>>>(emb, decin, hprev, xh, t);

    // h = tanh([x,h] @ [W_ih|W_hh]^T + (b_ih+b_hh)) -> f32 + bf16 copies
    gemm_wmma_kernel<512, true, true, false, true><<<4, 256, 0, stream>>>(
        xh, 512, Wcat_bf, bcat, hnext, h_bf, 256, 0, 2, 16);

    // hV = h @ [Vm;Vc]^T : [32,256]x[512,256]^T -> [32,512]
    gemm_wmma_kernel<256, false, false, false, false><<<8, 256, 0, stream>>>(
        h_bf, 256, Vcat_bf, nullptr, hV, nullptr, 512, 0, 2, 32);

    // energies (tanh-reduce), both heads in one pass
    energy_kernel<<<2048, 256, 0, stream>>>(encWm, encWc, hV, weffm, weffc,
                                            vbm, rm, vbc, rc, e_m, e_c);

    // alpha recurrence + chunkwise beta (per-batch LDS scans)
    alphabeta_kernel<<<BQ, TENC, 0, stream>>>(e_m, e_c,
                                              noise + (size_t)t * BQ * TENC,
                                              alpha, betab);

    // context = beta @ enc ; cat(bf16) = [context, h]
    context_kernel<<<BQ, 256, 0, stream>>>(betab, enc, h_bf, cat_bf);

    // attn = tanh(cat @ W_comb^T) -> bf16 : [32,512]x[256,512]^T
    gemm_wmma_kernel<512, false, true, true, false><<<4, 256, 0, stream>>>(
        cat_bf, 512, Wcomb_bf, nullptr, attn_bf, nullptr, 256, 0, 2, 16);

    // logits = attn @ W_proj^T + b_proj -> out[:, t, :] : [32,256]x[32000,256]^T
    gemm_wmma_kernel<256, true, false, false, false><<<500, 256, 0, stream>>>(
        attn_bf, 256, Wproj_bf, bproj, out, nullptr,
        (long)TDEC * NVOC, (long)t * NVOC, 2, NVOC / 16);

    float* tmp = hprev; hprev = hnext; hnext = tmp;
  }
}
